// BinLinear_81389630259625
// MI455X (gfx1250) — matrix operand order repaired and checked by reading, not compiled, not run on hardware
//
#include <hip/hip_runtime.h>

// CDNA5 / gfx1250 WMMA types (wave32)
typedef __attribute__((ext_vector_type(16))) __bf16 v16bf;
typedef __attribute__((ext_vector_type(8)))  __bf16 v8bf;
typedef __attribute__((ext_vector_type(4)))  __bf16 v4bf;
typedef __attribute__((ext_vector_type(8)))  float  v8f;
typedef __attribute__((ext_vector_type(4)))  int    v4i;

#define BM   128
#define BN   128
#define BK   32
#define LDK  40      // padded LDS row stride (bf16 elems): 80B = 20 banks -> conflict-free b128 reads
#define KDIM 2048
#define NDIM 2048

// --------------------------------------------------------------------------
// Async global->LDS support (CDNA5). Prototype (from compiler diagnostic):
//   void __builtin_amdgcn_global_load_async_to_lds_b128(
//       v4i AS1* gsrc, v4i AS3* ldsdst, imm int offset, imm int cpol)
// --------------------------------------------------------------------------
#if defined(__has_builtin)
#  if __has_builtin(__builtin_amdgcn_global_load_async_to_lds_b128)
#    define USE_ASYNC_LDS 1
#  endif
#endif
#ifndef USE_ASYNC_LDS
#  define USE_ASYNC_LDS 0
#endif

#if USE_ASYNC_LDS
typedef __attribute__((address_space(1))) v4i as1_v4i;
typedef __attribute__((address_space(3))) v4i as3_v4i;

__device__ __forceinline__ void async_b128(const void* g, void* l) {
    __builtin_amdgcn_global_load_async_to_lds_b128(
        (as1_v4i*)g, (as3_v4i*)l, /*offset=*/0, /*cpol=*/0);
}
__device__ __forceinline__ void wait_async0() {
#if __has_builtin(__builtin_amdgcn_s_wait_asynccnt)
    __builtin_amdgcn_s_wait_asynccnt(0);
#else
    asm volatile("s_wait_asynccnt 0" ::: "memory");
#endif
}
#endif

// ---------------------------------------------------------------------------
// Kernel 1: split fp32 input into bf16 hi + bf16 lo planes (one pass).
// ---------------------------------------------------------------------------
__global__ __launch_bounds__(256)
void split_hi_lo(const float* __restrict__ A, __bf16* __restrict__ hi,
                 __bf16* __restrict__ lo)
{
    const size_t i = (size_t)blockIdx.x * blockDim.x + threadIdx.x;  // float4 index
    const float4 v = ((const float4*)A)[i];
    const float vv[4] = {v.x, v.y, v.z, v.w};
    v4bf h, l;
    #pragma unroll
    for (int e = 0; e < 4; ++e) {
        const __bf16 he = (__bf16)vv[e];
        h[e] = he;
        l[e] = (__bf16)(vv[e] - (float)he);
    }
    *(v4bf*)(hi + i * 4) = h;
    *(v4bf*)(lo + i * 4) = l;
}

// ---------------------------------------------------------------------------
// Kernel 2: binarize weight (tanh(w)>=0 <=> w>=0) and transpose to Wt[n][k].
// ---------------------------------------------------------------------------
__global__ __launch_bounds__(256)
void binarize_transpose(const float* __restrict__ W, __bf16* __restrict__ Wt)
{
    __shared__ __bf16 tile[64][72];
    const int tid = threadIdx.x;
    const int k0  = blockIdx.y * 64;
    const int n0  = blockIdx.x * 64;

    #pragma unroll
    for (int j = 0; j < 4; ++j) {
        const int c  = tid + 256 * j;
        const int r  = c >> 4;
        const int c4 = (c & 15) << 2;
        const float4 v = *(const float4*)(W + (size_t)(k0 + r) * NDIM + n0 + c4);
        const float vv[4] = {v.x, v.y, v.z, v.w};
        #pragma unroll
        for (int e = 0; e < 4; ++e)
            tile[r][c4 + e] = (vv[e] >= 0.0f) ? (__bf16)1.0f : (__bf16)-1.0f;
    }
    __syncthreads();
    #pragma unroll
    for (int j = 0; j < 2; ++j) {
        const int c  = tid + 256 * j;
        const int n  = c >> 3;
        const int kc = (c & 7) << 3;
        v8bf o;
        #pragma unroll
        for (int e = 0; e < 8; ++e) o[e] = tile[kc + e][n];
        *(v8bf*)(Wt + (size_t)(n0 + n) * KDIM + k0 + kc) = o;
    }
}

// ---------------------------------------------------------------------------
// Kernel 3: pure-bf16 double-buffered WMMA GEMM.  out = (hi+lo) @ Wt^T
// ---------------------------------------------------------------------------
__global__ __launch_bounds__(256)
void gemm_bf16_wmma(const __bf16* __restrict__ Ahi, const __bf16* __restrict__ Alo,
                    const __bf16* __restrict__ Wt, float* __restrict__ C)
{
    __shared__ __align__(16) __bf16 sAh[2][BM * LDK];
    __shared__ __align__(16) __bf16 sAl[2][BM * LDK];
    __shared__ __align__(16) __bf16 sB [2][BN * LDK];   // [n][k]

    const int tid  = threadIdx.x;
    const int lane = tid & 31;
    const int wave = tid >> 5;      // 0..7
    const int half = lane >> 4;
    const int l16  = lane & 15;

    const int wave_m = wave >> 2;   // 0..1 -> row offset 64*wave_m
    const int wave_n = wave & 3;    // 0..3 -> col offset 32*wave_n

    const int m0 = blockIdx.y * BM;
    const int n0 = blockIdx.x * BN;

    v8f acc[4][2];
    #pragma unroll
    for (int i = 0; i < 4; ++i)
        #pragma unroll
        for (int j = 0; j < 2; ++j)
            #pragma unroll
            for (int e = 0; e < 8; ++e)
                acc[i][j][e] = 0.0f;

    // per-thread constant offsets (32-bit; bf16 planes are <= 33 MB)
    unsigned aoff[2], woff[2], soff[2];
    #pragma unroll
    for (int j = 0; j < 2; ++j) {
        const unsigned c   = (unsigned)tid + 256u * j;   // 0..511 -> 128 rows x 4 chunks
        const unsigned row = c >> 2;
        const unsigned kc  = (c & 3u) << 3;
        aoff[j] = (unsigned)(m0 + row) * KDIM + kc;
        woff[j] = (unsigned)(n0 + row) * KDIM + kc;
        soff[j] = row * LDK + kc;
    }

    const int NKT = KDIM / BK;   // 64

    // ---- shared compute section (reads LDS buffer `buf`) ------------------
    auto compute = [&](int buf) {
        v16bf bfrag[2];
        #pragma unroll
        for (int nt = 0; nt < 2; ++nt) {
            const __bf16* bp = &sB[buf][(wave_n * 32 + nt * 16 + l16) * LDK + half * 8];
            const v8bf b0 = *(const v8bf*)(bp);
            const v8bf b1 = *(const v8bf*)(bp + 16);
            #pragma unroll
            for (int e = 0; e < 8; ++e) { bfrag[nt][e] = b0[e]; bfrag[nt][8 + e] = b1[e]; }
        }
        #pragma unroll
        for (int mt = 0; mt < 4; ++mt) {
            const int row = wave_m * 64 + mt * 16 + l16;
            const __bf16* ah = &sAh[buf][row * LDK];
            const __bf16* al = &sAl[buf][row * LDK];
            const v8bf h0 = *(const v8bf*)(ah + half * 8);
            const v8bf h1 = *(const v8bf*)(ah + 16 + half * 8);
            const v8bf l0 = *(const v8bf*)(al + half * 8);
            const v8bf l1 = *(const v8bf*)(al + 16 + half * 8);
            v16bf ahi, alo;
            #pragma unroll
            for (int e = 0; e < 8; ++e) {
                ahi[e] = h0[e]; ahi[8 + e] = h1[e];
                alo[e] = l0[e]; alo[8 + e] = l1[e];
            }
            #pragma unroll
            for (int nt = 0; nt < 2; ++nt) {
                acc[mt][nt] = __builtin_amdgcn_wmma_f32_16x16x32_bf16(
                    false, ahi, false, bfrag[nt], (short)0, acc[mt][nt], false, false);
                acc[mt][nt] = __builtin_amdgcn_wmma_f32_16x16x32_bf16(
                    false, alo, false, bfrag[nt], (short)0, acc[mt][nt], false, false);
            }
        }
    };

#if USE_ASYNC_LDS
    // ---- async pipeline: no data VGPRs; hardware overlaps copy with WMMA --
    auto issue_tile = [&](int kt, int buf) {
        const unsigned kb = (unsigned)kt * BK;
        #pragma unroll
        for (int j = 0; j < 2; ++j) {
            async_b128(Ahi + (aoff[j] + kb), &sAh[buf][soff[j]]);
            async_b128(Alo + (aoff[j] + kb), &sAl[buf][soff[j]]);
            async_b128(Wt  + (woff[j] + kb), &sB [buf][soff[j]]);
        }
    };
    issue_tile(0, 0);
    wait_async0();
    __syncthreads();
    for (int kt = 0; kt < NKT; ++kt) {
        const int buf = kt & 1;
        if (kt + 1 < NKT) issue_tile(kt + 1, buf ^ 1);  // async copy runs during WMMA
        compute(buf);
        if (kt + 1 < NKT) wait_async0();
        __syncthreads();
    }
#else
    // ---- register staging with SHORT live ranges (no cross-WMMA liveness) -
    uint4 rh[2], rl[2], rw[2];
    auto load_regs = [&](int kt) {
        const unsigned kb = (unsigned)kt * BK;
        #pragma unroll
        for (int j = 0; j < 2; ++j) {
            rh[j] = *(const uint4*)(Ahi + (aoff[j] + kb));
            rl[j] = *(const uint4*)(Alo + (aoff[j] + kb));
            rw[j] = *(const uint4*)(Wt  + (woff[j] + kb));
        }
    };
    auto stage = [&](int buf) {
        #pragma unroll
        for (int j = 0; j < 2; ++j) {
            *(uint4*)(&sAh[buf][soff[j]]) = rh[j];
            *(uint4*)(&sAl[buf][soff[j]]) = rl[j];
            *(uint4*)(&sB [buf][soff[j]]) = rw[j];
        }
    };
    load_regs(0);
    stage(0);
    __syncthreads();
    for (int kt = 0; kt < NKT; ++kt) {
        const int buf = kt & 1;
        compute(buf);
        if (kt + 1 < NKT) {
            load_regs(kt + 1);
            stage(buf ^ 1);
        }
        __syncthreads();
    }
#endif

    // epilogue
    #pragma unroll
    for (int mt = 0; mt < 4; ++mt)
        #pragma unroll
        for (int nt = 0; nt < 2; ++nt) {
            const unsigned row0 = (unsigned)(m0 + wave_m * 64 + mt * 16 + half * 8);
            const unsigned col  = (unsigned)(n0 + wave_n * 32 + nt * 16 + l16);
            #pragma unroll
            for (int r = 0; r < 8; ++r)
                C[(size_t)((row0 + r) * (unsigned)NDIM + col)] = acc[mt][nt][r];
        }
}

// ---------------------------------------------------------------------------
// Fallback: fused single-pass kernel (used only if workspace is too small).
// ---------------------------------------------------------------------------
__global__ __launch_bounds__(256)
void binlinear_fused_fallback(const float* __restrict__ A,
                              const float* __restrict__ W,
                              float* __restrict__ C)
{
    __shared__ __align__(16) __bf16 As_hi[BM * LDK];
    __shared__ __align__(16) __bf16 As_lo[BM * LDK];
    __shared__ __align__(16) __bf16 Bs[BN * LDK];

    const int tid  = threadIdx.x;
    const int lane = tid & 31;
    const int wave = tid >> 5;
    const int half = lane >> 4;
    const int l16  = lane & 15;
    const int wave_m = wave >> 2;
    const int wave_n = wave & 3;
    const int m0 = blockIdx.y * BM;
    const int n0 = blockIdx.x * BN;

    v8f acc[4][2];
    #pragma unroll
    for (int i = 0; i < 4; ++i)
        #pragma unroll
        for (int j = 0; j < 2; ++j)
            #pragma unroll
            for (int e = 0; e < 8; ++e)
                acc[i][j][e] = 0.0f;

    auto load_stage = [&](int kt) {
        const unsigned kb = (unsigned)kt * BK;
        #pragma unroll
        for (int j = 0; j < 4; ++j) {
            const unsigned i    = (unsigned)tid + 256u * j;
            const unsigned arow = i >> 3;
            const unsigned acol = (i & 7u) << 2;
            const float4 av = *(const float4*)(A + ((unsigned)(m0 + arow) * KDIM + kb + acol));
            const float avv[4] = {av.x, av.y, av.z, av.w};
            #pragma unroll
            for (int e = 0; e < 4; ++e) {
                const float  a = avv[e];
                const __bf16 h = (__bf16)a;
                const __bf16 l = (__bf16)(a - (float)h);
                As_hi[arow * LDK + acol + e] = h;
                As_lo[arow * LDK + acol + e] = l;
            }
            const unsigned krow = i >> 5;
            const unsigned ncol = (i & 31u) << 2;
            const float4 wv = *(const float4*)(W + ((kb + krow) * (unsigned)NDIM + n0 + ncol));
            const float wvv[4] = {wv.x, wv.y, wv.z, wv.w};
            #pragma unroll
            for (int e = 0; e < 4; ++e) {
                const __bf16 pb = (wvv[e] >= 0.0f) ? (__bf16)1.0f : (__bf16)-1.0f;
                Bs[(ncol + e) * LDK + krow] = pb;
            }
        }
    };

    const int NKT = KDIM / BK;
    load_stage(0);
    __syncthreads();

    for (int kt = 0; kt < NKT; ++kt) {
        v16bf bfrag[2];
        #pragma unroll
        for (int nt = 0; nt < 2; ++nt) {
            const __bf16* bp = &Bs[(wave_n * 32 + nt * 16 + l16) * LDK + half * 8];
            const v8bf b0 = *(const v8bf*)(bp);
            const v8bf b1 = *(const v8bf*)(bp + 16);
            #pragma unroll
            for (int e = 0; e < 8; ++e) { bfrag[nt][e] = b0[e]; bfrag[nt][8 + e] = b1[e]; }
        }
        #pragma unroll
        for (int mt = 0; mt < 4; ++mt) {
            const int row = wave_m * 64 + mt * 16 + l16;
            const __bf16* ah = &As_hi[row * LDK];
            const __bf16* al = &As_lo[row * LDK];
            const v8bf h0 = *(const v8bf*)(ah + half * 8);
            const v8bf h1 = *(const v8bf*)(ah + 16 + half * 8);
            const v8bf l0 = *(const v8bf*)(al + half * 8);
            const v8bf l1 = *(const v8bf*)(al + 16 + half * 8);
            v16bf ahi, alo;
            #pragma unroll
            for (int e = 0; e < 8; ++e) {
                ahi[e] = h0[e]; ahi[8 + e] = h1[e];
                alo[e] = l0[e]; alo[8 + e] = l1[e];
            }
            #pragma unroll
            for (int nt = 0; nt < 2; ++nt) {
                acc[mt][nt] = __builtin_amdgcn_wmma_f32_16x16x32_bf16(
                    false, ahi, false, bfrag[nt], (short)0, acc[mt][nt], false, false);
                acc[mt][nt] = __builtin_amdgcn_wmma_f32_16x16x32_bf16(
                    false, alo, false, bfrag[nt], (short)0, acc[mt][nt], false, false);
            }
        }
        __syncthreads();
        if (kt + 1 < NKT) {
            load_stage(kt + 1);
            __syncthreads();
        }
    }

    #pragma unroll
    for (int mt = 0; mt < 4; ++mt)
        #pragma unroll
        for (int nt = 0; nt < 2; ++nt) {
            const unsigned row0 = (unsigned)(m0 + wave_m * 64 + mt * 16 + half * 8);
            const unsigned col  = (unsigned)(n0 + wave_n * 32 + nt * 16 + l16);
            #pragma unroll
            for (int r = 0; r < 8; ++r)
                C[(size_t)((row0 + r) * (unsigned)NDIM + col)] = acc[mt][nt][r];
        }
}

// ---------------------------------------------------------------------------
extern "C" void kernel_launch(void* const* d_in, const int* in_sizes, int n_in,
                              void* d_out, int out_size, void* d_ws, size_t ws_size,
                              hipStream_t stream) {
    (void)n_in; (void)out_size;
    const float* inp = (const float*)d_in[0];   // [M, 2048] fp32
    const float* wgt = (const float*)d_in[1];   // [2048, 2048] fp32
    float* out = (float*)d_out;                 // [M, 2048] fp32

    const int M = in_sizes[0] / KDIM;           // 8192
    const size_t elemsA = (size_t)M * KDIM;
    const size_t elemsW = (size_t)KDIM * NDIM;
    const size_t need   = (2 * elemsA + elemsW) * sizeof(__bf16);  // 72 MB

    dim3 gemm_grid(NDIM / BN, M / BM);

    if (ws_size >= need) {
        __bf16* hi = (__bf16*)d_ws;
        __bf16* lo = hi + elemsA;
        __bf16* wt = lo + elemsA;

        split_hi_lo<<<(unsigned)(elemsA / 4 / 256), 256, 0, stream>>>(inp, hi, lo);
        binarize_transpose<<<dim3(NDIM / 64, KDIM / 64), 256, 0, stream>>>(wgt, wt);
        gemm_bf16_wmma<<<gemm_grid, 256, 0, stream>>>(hi, lo, wt, out);
    } else {
        binlinear_fused_fallback<<<gemm_grid, 256, 0, stream>>>(inp, wgt, out);
    }
}
